// IterativeCrossAttTransformer_28458453303816
// MI455X (gfx1250) — compile-verified, matrix operand order checked by reading, not run on hardware
//
#include <hip/hip_runtime.h>
#include <hip/hip_bf16.h>
#include <math.h>

typedef __bf16 bf16;
typedef __attribute__((ext_vector_type(16))) __bf16 v16bf;
typedef __attribute__((ext_vector_type(8)))  float  v8f;
typedef __attribute__((ext_vector_type(4)))  unsigned int v4u;
typedef __attribute__((ext_vector_type(8)))  unsigned int v8u;

#define DMODEL 1024
#define MTOK   4096          // B*N = 8*512 token rows per stream
#define NEGINF (-1.0e9f)

__device__ __forceinline__ bf16 f2bf(float v) { return (bf16)v; }

// ---------------------------------------------------------------------------
// CDNA5 16-bit A/B WMMA fragment from a bf16 row-major buffer.
// lane l<16 -> row=l, K = {kb..kb+7, kb+16..kb+23} with kb=0;
// lane l>=16 -> same row (pass row incl. lane&15), kb=8.
// ---------------------------------------------------------------------------
__device__ __forceinline__ v16bf load_frag_rows(const bf16* base, int stride,
                                                int row, int half)
{
    const bf16* p = base + (size_t)row * stride + half * 8;
    v16bf f;
#pragma unroll
    for (int i = 0; i < 8; ++i) { f[i] = p[i]; f[8 + i] = p[16 + i]; }
    return f;
}

// ---------------------------------------------------------------------------
// Tensor Data Mover: DMA a [tileRows x tileCols] bf16 tile (row stride
// rowStrideElems in global) into LDS at ldsByte, padding 4 DWORDs after each
// 16 DWORDs (64B row) -> LDS row stride 80B (= 40 bf16, bank-conflict-free).
// D# per CDNA5 ISA. Issued via inline asm so it
// is portable across ROCm-7.2 / amdgpu-toolchain builtin arities.
// ---------------------------------------------------------------------------
__device__ __forceinline__ void tdm_load_2d(const bf16* g, unsigned ldsByte,
                                            int tileRows, int tileCols,
                                            int rowStrideElems, int tensorRows)
{
    const unsigned long long ga = (unsigned long long)(size_t)g;
    v4u g0; v8u g1;
    g0[0] = 1u;                                           // count=1 (user D#)
    g0[1] = ldsByte;                                      // lds_addr
    g0[2] = (unsigned)ga;                                 // global_addr[31:0]
    g0[3] = (unsigned)((ga >> 32) & 0x01FFFFFFull)        // global_addr[56:32]
          | (2u << 30);                                   // type=2 ("image")
    g1[0] = (1u << 16)                                    // data_size: 2 bytes
          | (1u << 20)                                    // pad_enable
          | (3u << 22)                                    // pad_interval: 16 DW
          | (3u << 25);                                   // pad_amount: 4 DW
    g1[1] = ((unsigned)rowStrideElems & 0xFFFFu) << 16;   // tensor_dim0[15:0]
    g1[2] = (((unsigned)rowStrideElems >> 16) & 0xFFFFu)  // tensor_dim0[31:16]
          | (((unsigned)tensorRows & 0xFFFFu) << 16);     // tensor_dim1[15:0]
    g1[3] = (((unsigned)tensorRows >> 16) & 0xFFFFu)      // tensor_dim1[31:16]
          | (((unsigned)tileCols & 0xFFFFu) << 16);       // tile_dim0
    g1[4] = ((unsigned)tileRows & 0xFFFFu);               // tile_dim1 (dim2=0)
    g1[5] = (unsigned)rowStrideElems;                     // dim0_stride[31:0]
    g1[6] = 0u;                                           // stride hi, dim1_stride
    g1[7] = 0u;
    asm volatile("tensor_load_to_lds %0, %1" :: "s"(g0), "s"(g1) : "memory");
}

// ---------------------------------------------------------------------------
// GEMM: out = act(A[M,K] @ W[N,K]^T + bias[N]) (+ resid[M,N])
// 256 threads (8 waves). Block tile 128x128, wave tile 32x64, K-step 32.
// Tiles staged by TDM, double-buffered; 8 WMMAs / wave / K-step.
// ---------------------------------------------------------------------------
#define BM 128
#define BN 128
#define BK 32
#define LDS_STRIDE 40

template <int ACT>
__global__ __launch_bounds__(256)
void gemm_bf16_kernel(const bf16* __restrict__ A, const bf16* __restrict__ W,
                      const float* __restrict__ bias, const float* __restrict__ resid,
                      float* __restrict__ outf, bf16* __restrict__ outb,
                      int M, int N, int K)
{
    __shared__ bf16 As[2][BM * LDS_STRIDE];
    __shared__ bf16 Bs[2][BN * LDS_STRIDE];

    const int t    = threadIdx.x;
    const int lane = t & 31;
    const int wave = t >> 5;
    const int half = lane >> 4;
    const int l15  = lane & 15;
    const int mw   = wave >> 1;   // 0..3  (rows of 32)
    const int nw   = wave & 1;    // 0..1  (cols of 64)
    const int mBase = blockIdx.y * BM;
    const int nBase = blockIdx.x * BN;

    // scalar-uniform issuer flag -> s_cbranch skips TDM region entirely
    const bool issuer = (__builtin_amdgcn_readfirstlane(wave) == 0);

    const v8f vzero = {};
    v8f c[2][4];
#pragma unroll
    for (int i = 0; i < 2; ++i)
#pragma unroll
        for (int j = 0; j < 4; ++j) c[i][j] = vzero;

    if (issuer) {
        tdm_load_2d(A + (size_t)mBase * K, (unsigned)(size_t)&As[0][0], BM, BK, K, M);
        tdm_load_2d(W + (size_t)nBase * K, (unsigned)(size_t)&Bs[0][0], BN, BK, K, N);
        asm volatile("s_wait_tensorcnt 0x0" ::: "memory");
    }
    __syncthreads();

    int cur = 0;
    for (int k0 = 0; k0 < K; k0 += BK) {
        if (issuer && (k0 + BK < K)) {   // prefetch next tiles into other buffer
            tdm_load_2d(A + (size_t)mBase * K + k0 + BK,
                        (unsigned)(size_t)&As[cur ^ 1][0], BM, BK, K, M);
            tdm_load_2d(W + (size_t)nBase * K + k0 + BK,
                        (unsigned)(size_t)&Bs[cur ^ 1][0], BN, BK, K, N);
        }

        v16bf af[2], bfr[4];
#pragma unroll
        for (int i = 0; i < 2; ++i)
            af[i] = load_frag_rows(&As[cur][0], LDS_STRIDE, mw * 32 + i * 16 + l15, half);
#pragma unroll
        for (int j = 0; j < 4; ++j)
            bfr[j] = load_frag_rows(&Bs[cur][0], LDS_STRIDE, nw * 64 + j * 16 + l15, half);

#pragma unroll
        for (int i = 0; i < 2; ++i)
#pragma unroll
            for (int j = 0; j < 4; ++j)
                c[i][j] = __builtin_amdgcn_wmma_f32_16x16x32_bf16(
                    false, af[i], false, bfr[j], (short)0, c[i][j], false, false);

        if (issuer) asm volatile("s_wait_tensorcnt 0x0" ::: "memory");
        __syncthreads();
        cur ^= 1;
    }

    // epilogue: C layout -> element (M = r + 8*half, N = lane&15)
#pragma unroll
    for (int i = 0; i < 2; ++i) {
#pragma unroll
        for (int j = 0; j < 4; ++j) {
            const int ncol = nBase + nw * 64 + j * 16 + l15;
            const float bv = bias ? bias[ncol] : 0.0f;
#pragma unroll
            for (int r = 0; r < 8; ++r) {
                const int mrow = mBase + mw * 32 + i * 16 + r + 8 * half;
                float v = c[i][j][r] + bv;
                if (ACT == 1) v = fmaxf(v, 0.0f);
                if (ACT == 2) v = 0.5f * v * (1.0f + erff(v * 0.70710678118f));
                const size_t idx = (size_t)mrow * N + ncol;
                if (resid) v += resid[idx];
                if (outf) outf[idx] = v;
                if (outb) outb[idx] = f2bf(v);
            }
        }
    }
}

// ---------------------------------------------------------------------------
// LayerNorm over D=1024, one row per block (256 threads x float4).
// ---------------------------------------------------------------------------
__global__ __launch_bounds__(256)
void ln_kernel(const float* __restrict__ x, const float* __restrict__ g,
               const float* __restrict__ b, float* __restrict__ outf,
               bf16* __restrict__ outb)
{
    __shared__ float red[256];
    const int row = blockIdx.x;
    const int t = threadIdx.x;
    const float4 xv = ((const float4*)(x + (size_t)row * DMODEL))[t];

    red[t] = xv.x + xv.y + xv.z + xv.w;
    __syncthreads();
    for (int o = 128; o > 0; o >>= 1) { if (t < o) red[t] += red[t + o]; __syncthreads(); }
    const float mean = red[0] * (1.0f / DMODEL);
    __syncthreads();

    const float dx = xv.x - mean, dy = xv.y - mean, dz = xv.z - mean, dw = xv.w - mean;
    red[t] = dx * dx + dy * dy + dz * dz + dw * dw;
    __syncthreads();
    for (int o = 128; o > 0; o >>= 1) { if (t < o) red[t] += red[t + o]; __syncthreads(); }
    const float rstd = rsqrtf(red[0] * (1.0f / DMODEL) + 1e-5f);

    const int c0 = t * 4;
    const float y0 = dx * rstd * g[c0 + 0] + b[c0 + 0];
    const float y1 = dy * rstd * g[c0 + 1] + b[c0 + 1];
    const float y2 = dz * rstd * g[c0 + 2] + b[c0 + 2];
    const float y3 = dw * rstd * g[c0 + 3] + b[c0 + 3];
    const size_t base = (size_t)row * DMODEL + c0;
    if (outf) { outf[base] = y0; outf[base + 1] = y1; outf[base + 2] = y2; outf[base + 3] = y3; }
    if (outb) { outb[base] = f2bf(y0); outb[base + 1] = f2bf(y1);
                outb[base + 2] = f2bf(y2); outb[base + 3] = f2bf(y3); }
}

// ---------------------------------------------------------------------------
// GRU gate fusion: gi/gh are [M, 3*D] (r,z,n gates), h updated in place.
// ---------------------------------------------------------------------------
__global__ __launch_bounds__(256)
void gru_gate_kernel(const float* __restrict__ gi, const float* __restrict__ gh,
                     float* __restrict__ h, bf16* __restrict__ hb)
{
    const size_t idx = (size_t)blockIdx.x * blockDim.x + threadIdx.x;
    const size_t r = idx >> 10, c = idx & 1023;
    const size_t base = r * (3 * DMODEL) + c;
    const float ir = gi[base], iz = gi[base + DMODEL], in_ = gi[base + 2 * DMODEL];
    const float hr = gh[base], hz = gh[base + DMODEL], hn = gh[base + 2 * DMODEL];
    const float rg = 1.0f / (1.0f + __expf(-(ir + hr)));
    const float zg = 1.0f / (1.0f + __expf(-(iz + hz)));
    const float ng = tanhf(in_ + rg * hn);
    const float hp = h[idx];
    const float out = (1.0f - zg) * ng + zg * hp;
    h[idx] = out;
    hb[idx] = f2bf(out);
}

__global__ __launch_bounds__(256)
void add_kernel(const float* __restrict__ a, const float* __restrict__ b,
                float* __restrict__ out)
{
    const size_t idx = (size_t)blockIdx.x * blockDim.x + threadIdx.x;
    out[idx] = a[idx] + b[idx];
}

// concat write-out: src [4096,1024] fp32 -> out[b, sel*512 + n, :]
__global__ __launch_bounds__(256)
void writeout_kernel(const float* __restrict__ src, float* __restrict__ out, int sel)
{
    const size_t idx = (size_t)blockIdx.x * blockDim.x + threadIdx.x;
    const size_t row = idx >> 10, col = idx & 1023;
    const size_t bb = row >> 9, n = row & 511;
    out[((bb << 10) + (size_t)sel * 512 + n) * DMODEL + col] = src[idx];
}

__global__ __launch_bounds__(256)
void cast_kernel(const float* __restrict__ src, bf16* __restrict__ dst, int n)
{
    const int i = blockIdx.x * blockDim.x + threadIdx.x;
    if (i < n) dst[i] = f2bf(src[i]);
}

// ---------------------------------------------------------------------------
// Flash-style cross attention: one wave per 16-row q tile over 512 keys.
// q,k,v merged-layout bf16: token row stride 1024, head offset h*64.
// ---------------------------------------------------------------------------
__global__ __launch_bounds__(256)
void attn_kernel(const bf16* __restrict__ q, const bf16* __restrict__ kk,
                 const bf16* __restrict__ vv, const int* __restrict__ maskq,
                 const int* __restrict__ maskk, bf16* __restrict__ out)
{
    __shared__ bf16 sp[8][16][36];   // per-wave 16x32 probability tile (+pad)

    const int lane = threadIdx.x & 31;
    const int wave = threadIdx.x >> 5;
    const int half = lane >> 4;
    const int l15  = lane & 15;
    const int tile = blockIdx.x * 8 + wave;   // 0..4095
    const int qt = tile & 31;
    const int bh = tile >> 5;
    const int b  = bh >> 4;
    const int h  = bh & 15;
    const int n0 = qt * 16;
    const size_t hoff = (size_t)h * 64;

    // q fragments: 16 rows x 64 -> two K=32 A-fragments
    v16bf qf[2];
#pragma unroll
    for (int kc = 0; kc < 2; ++kc) {
        const bf16* p = q + ((size_t)(b * 512 + n0 + l15)) * DMODEL + hoff + kc * 32 + half * 8;
        v16bf f;
#pragma unroll
        for (int i = 0; i < 8; ++i) { f[i] = p[i]; f[8 + i] = p[16 + i]; }
        qf[kc] = f;
    }

    int mq[8];
#pragma unroll
    for (int e = 0; e < 8; ++e) mq[e] = maskq[b * 512 + n0 + e + 8 * half];

    const v8f vzero = {};
    v8f acc[4];
#pragma unroll
    for (int d = 0; d < 4; ++d) acc[d] = vzero;
    float mrun[8], lrun[8];
#pragma unroll
    for (int e = 0; e < 8; ++e) { mrun[e] = -3.0e38f; lrun[e] = 0.0f; }

    for (int mc = 0; mc < 512; mc += 32) {
        if (mc + 32 < 512)
            __builtin_prefetch(kk + ((size_t)(b * 512 + mc + 32 + l15)) * DMODEL + hoff, 0, 0);

        v8f s[2]; s[0] = vzero; s[1] = vzero;
#pragma unroll
        for (int tj = 0; tj < 2; ++tj)
#pragma unroll
            for (int kc = 0; kc < 2; ++kc) {
                const bf16* p = kk + ((size_t)(b * 512 + mc + tj * 16 + l15)) * DMODEL
                                + hoff + kc * 32 + half * 8;
                v16bf f;
#pragma unroll
                for (int i = 0; i < 8; ++i) { f[i] = p[i]; f[8 + i] = p[16 + i]; }
                s[tj] = __builtin_amdgcn_wmma_f32_16x16x32_bf16(
                    false, qf[kc], false, f, (short)0, s[tj], false, false);
            }

        const int mk0 = maskk[b * 512 + mc + l15];
        const int mk1 = maskk[b * 512 + mc + 16 + l15];
#pragma unroll
        for (int e = 0; e < 8; ++e) {
            const float s0 = s[0][e] * 0.125f;   // DH^-0.5
            const float s1 = s[1][e] * 0.125f;
            s[0][e] = (mq[e] && mk0) ? s0 : NEGINF;
            s[1][e] = (mq[e] && mk1) ? s1 : NEGINF;
        }

        // online softmax, per-row reductions across the 16-lane half
#pragma unroll
        for (int e = 0; e < 8; ++e) {
            float vm = fmaxf(s[0][e], s[1][e]);
#pragma unroll
            for (int o = 1; o < 16; o <<= 1) vm = fmaxf(vm, __shfl_xor(vm, o, 32));
            const float mn = fmaxf(mrun[e], vm);
            const float corr = __expf(mrun[e] - mn);
            mrun[e] = mn;
            const float p0 = __expf(s[0][e] - mn);
            const float p1 = __expf(s[1][e] - mn);
            s[0][e] = p0; s[1][e] = p1;
            float rs = p0 + p1;
#pragma unroll
            for (int o = 1; o < 16; o <<= 1) rs += __shfl_xor(rs, o, 32);
            lrun[e] = lrun[e] * corr + rs;
#pragma unroll
            for (int d = 0; d < 4; ++d) acc[d][e] *= corr;
        }

        // C-layout p tile -> LDS -> A-fragment layout
#pragma unroll
        for (int tj = 0; tj < 2; ++tj)
#pragma unroll
            for (int e = 0; e < 8; ++e)
                sp[wave][e + 8 * half][tj * 16 + l15] = f2bf(s[tj][e]);
        asm volatile("s_wait_dscnt 0x0" ::: "memory");

        v16bf pf;
        {
            const bf16* p = &sp[wave][l15][half * 8];
#pragma unroll
            for (int i = 0; i < 8; ++i) { pf[i] = p[i]; pf[8 + i] = p[16 + i]; }
        }

        // v B-fragments (columns of v strided by row stride 1024)
#pragma unroll
        for (int d = 0; d < 4; ++d) {
            v16bf vf;
#pragma unroll
            for (int i = 0; i < 8; ++i) {
                vf[i]     = vv[((size_t)(b * 512 + mc + half * 8 + i)) * DMODEL + hoff + d * 16 + l15];
                vf[8 + i] = vv[((size_t)(b * 512 + mc + 16 + half * 8 + i)) * DMODEL + hoff + d * 16 + l15];
            }
            acc[d] = __builtin_amdgcn_wmma_f32_16x16x32_bf16(
                false, pf, false, vf, (short)0, acc[d], false, false);
        }
        asm volatile("s_wait_dscnt 0x0" ::: "memory");
    }

#pragma unroll
    for (int d = 0; d < 4; ++d)
#pragma unroll
        for (int e = 0; e < 8; ++e) {
            const float ov = acc[d][e] / lrun[e];
            out[((size_t)(b * 512 + n0 + e + 8 * half)) * DMODEL + hoff + d * 16 + l15] = f2bf(ov);
        }
}

// ---------------------------------------------------------------------------
// Orchestration
// ---------------------------------------------------------------------------
extern "C" void kernel_launch(void* const* d_in, const int* in_sizes, int n_in,
                              void* d_out, int out_size, void* d_ws, size_t ws_size,
                              hipStream_t stream)
{
    (void)in_sizes; (void)n_in; (void)out_size; (void)ws_size;

    const float* x1 = (const float*)d_in[0];
    const float* x2 = (const float*)d_in[1];
    const int* mask1 = (const int*)d_in[2];
    const int* mask2 = (const int*)d_in[3];
    const float* norm1_g = (const float*)d_in[4];
    const float* norm1_b = (const float*)d_in[5];
    const float* norm2_g = (const float*)d_in[6];
    const float* norm2_b = (const float*)d_in[7];
    const float* n1a_g = (const float*)d_in[8];
    const float* n1a_b = (const float*)d_in[9];
    const float* n1b_g = (const float*)d_in[10];
    const float* n1b_b = (const float*)d_in[11];
    const float* n2a_g = (const float*)d_in[12];
    const float* n2a_b = (const float*)d_in[13];
    const float* n2b_g = (const float*)d_in[14];
    const float* n2b_b = (const float*)d_in[15];
    const float* wq = (const float*)d_in[16];
    const float* wk = (const float*)d_in[17];
    const float* wv = (const float*)d_in[18];
    const float* wp = (const float*)d_in[19];
    const float* b_proj = (const float*)d_in[20];
    const float* g1ih = (const float*)d_in[21];
    const float* g1hh = (const float*)d_in[22];
    const float* g1bih = (const float*)d_in[23];
    const float* g1bhh = (const float*)d_in[24];
    const float* g2ih = (const float*)d_in[25];
    const float* g2hh = (const float*)d_in[26];
    const float* g2bih = (const float*)d_in[27];
    const float* g2bhh = (const float*)d_in[28];
    const float* m1w1 = (const float*)d_in[29];
    const float* m1b1 = (const float*)d_in[30];
    const float* m1w2 = (const float*)d_in[31];
    const float* m1b2 = (const float*)d_in[32];
    const float* m2w1 = (const float*)d_in[33];
    const float* m2b1 = (const float*)d_in[34];
    const float* m2w2 = (const float*)d_in[35];
    const float* m2b2 = (const float*)d_in[36];
    const float* f1w = (const float*)d_in[37];
    const float* f1b = (const float*)d_in[38];
    const float* f2w = (const float*)d_in[39];
    const float* f2b = (const float*)d_in[40];

    char* ws = (char*)d_ws;
    size_t off = 0;
    auto alloc = [&](size_t bytes) -> void* {
        off = (off + 255) & ~(size_t)255;
        void* p = ws + off;
        off += bytes;
        return p;
    };

    const size_t MD = (size_t)MTOK * DMODEL;     // 4M elems
    const size_t W1 = (size_t)DMODEL * DMODEL;   // 1M
    const size_t W3 = 3 * W1;                    // 3M

    // bf16 weights
    bf16* wqB = (bf16*)alloc(W1 * 2);   bf16* wkB = (bf16*)alloc(W1 * 2);
    bf16* wvB = (bf16*)alloc(W1 * 2);   bf16* wpB = (bf16*)alloc(W1 * 2);
    bf16* g1ihB = (bf16*)alloc(W3 * 2); bf16* g1hhB = (bf16*)alloc(W3 * 2);
    bf16* g2ihB = (bf16*)alloc(W3 * 2); bf16* g2hhB = (bf16*)alloc(W3 * 2);
    bf16* m1w1B = (bf16*)alloc(W1 * 2); bf16* m1w2B = (bf16*)alloc(W1 * 2);
    bf16* m2w1B = (bf16*)alloc(W1 * 2); bf16* m2w2B = (bf16*)alloc(W1 * 2);
    bf16* f1B = (bf16*)alloc(W1 * 2);   bf16* f2B = (bf16*)alloc(W1 * 2);

    // activations
    float* a1f = (float*)alloc(MD * 4);  float* a2f = (float*)alloc(MD * 4);
    bf16* a1b = (bf16*)alloc(MD * 2);    bf16* a2b = (bf16*)alloc(MD * 2);
    bf16* ln1b = (bf16*)alloc(MD * 2);   bf16* ln2b = (bf16*)alloc(MD * 2);
    bf16* k1b = (bf16*)alloc(MD * 2);    bf16* v1b = (bf16*)alloc(MD * 2);
    bf16* k2b = (bf16*)alloc(MD * 2);    bf16* v2b = (bf16*)alloc(MD * 2);
    bf16* q1b = (bf16*)alloc(MD * 2);    bf16* q2b = (bf16*)alloc(MD * 2);
    bf16* at1b = (bf16*)alloc(MD * 2);   bf16* at2b = (bf16*)alloc(MD * 2);
    bf16* up1b = (bf16*)alloc(MD * 2);   bf16* up2b = (bf16*)alloc(MD * 2);
    bf16* midb = (bf16*)alloc(MD * 2);
    float* gif = (float*)alloc(3 * MD * 4);
    float* ghf = (float*)alloc(3 * MD * 4);
    float* x1of = (float*)alloc(MD * 4); float* x2of = (float*)alloc(MD * 4);
    float* tmpf = (float*)alloc(MD * 4);

    auto castw = [&](const float* s, bf16* d, int n) {
        cast_kernel<<<(n + 255) / 256, 256, 0, stream>>>(s, d, n);
    };
    auto gemm = [&](const bf16* A, const bf16* Wm, const float* bias, const float* resid,
                    float* of, bf16* ob, int N, int act) {
        dim3 g(N / BN, MTOK / BM), blk(256);
        if (act == 0)      gemm_bf16_kernel<0><<<g, blk, 0, stream>>>(A, Wm, bias, resid, of, ob, MTOK, N, DMODEL);
        else if (act == 1) gemm_bf16_kernel<1><<<g, blk, 0, stream>>>(A, Wm, bias, resid, of, ob, MTOK, N, DMODEL);
        else               gemm_bf16_kernel<2><<<g, blk, 0, stream>>>(A, Wm, bias, resid, of, ob, MTOK, N, DMODEL);
    };
    auto ln = [&](const float* x, const float* g, const float* bb, float* of, bf16* ob) {
        ln_kernel<<<MTOK, 256, 0, stream>>>(x, g, bb, of, ob);
    };
    const int EW_BLOCKS = (int)(MD / 256);

    // weight casts
    castw(wq, wqB, (int)W1); castw(wk, wkB, (int)W1);
    castw(wv, wvB, (int)W1); castw(wp, wpB, (int)W1);
    castw(g1ih, g1ihB, (int)W3); castw(g1hh, g1hhB, (int)W3);
    castw(g2ih, g2ihB, (int)W3); castw(g2hh, g2hhB, (int)W3);
    castw(m1w1, m1w1B, (int)W1); castw(m1w2, m1w2B, (int)W1);
    castw(m2w1, m2w1B, (int)W1); castw(m2w2, m2w2B, (int)W1);
    castw(f1w, f1B, (int)W1);    castw(f2w, f2B, (int)W1);

    // a = LN(x, norm1); static k/v
    ln(x1, norm1_g, norm1_b, a1f, a1b);
    ln(x2, norm1_g, norm1_b, a2f, a2b);
    gemm(a1b, wkB, nullptr, nullptr, nullptr, k1b, DMODEL, 0);
    gemm(a1b, wvB, nullptr, nullptr, nullptr, v1b, DMODEL, 0);
    gemm(a2b, wkB, nullptr, nullptr, nullptr, k2b, DMODEL, 0);
    gemm(a2b, wvB, nullptr, nullptr, nullptr, v2b, DMODEL, 0);

    for (int it = 0; it < 3; ++it) {
        // q projections
        ln(a1f, n1a_g, n1a_b, nullptr, ln1b);
        ln(a2f, n2a_g, n2a_b, nullptr, ln2b);
        gemm(ln1b, wqB, nullptr, nullptr, nullptr, q1b, DMODEL, 0);
        gemm(ln2b, wqB, nullptr, nullptr, nullptr, q2b, DMODEL, 0);

        // cross attention
        attn_kernel<<<512, 256, 0, stream>>>(q1b, k2b, v2b, mask1, mask2, at1b);
        attn_kernel<<<512, 256, 0, stream>>>(q2b, k1b, v1b, mask2, mask1, at2b);

        // output projection
        gemm(at1b, wpB, b_proj, nullptr, nullptr, up1b, DMODEL, 0);
        gemm(at2b, wpB, b_proj, nullptr, nullptr, up2b, DMODEL, 0);

        // GRU stream 1
        gemm(up1b, g1ihB, g1bih, nullptr, gif, nullptr, 3 * DMODEL, 0);
        gemm(a1b, g1hhB, g1bhh, nullptr, ghf, nullptr, 3 * DMODEL, 0);
        gru_gate_kernel<<<EW_BLOCKS, 256, 0, stream>>>(gif, ghf, a1f, a1b);
        // GRU stream 2
        gemm(up2b, g2ihB, g2bih, nullptr, gif, nullptr, 3 * DMODEL, 0);
        gemm(a2b, g2hhB, g2bhh, nullptr, ghf, nullptr, 3 * DMODEL, 0);
        gru_gate_kernel<<<EW_BLOCKS, 256, 0, stream>>>(gif, ghf, a2f, a2b);

        // MLPs with residual
        ln(a1f, n1b_g, n1b_b, nullptr, ln1b);
        gemm(ln1b, m1w1B, m1b1, nullptr, nullptr, midb, DMODEL, 1);
        gemm(midb, m1w2B, m1b2, a1f, a1f, a1b, DMODEL, 0);
        ln(a2f, n2b_g, n2b_b, nullptr, ln2b);
        gemm(ln2b, m2w1B, m2b1, nullptr, nullptr, midb, DMODEL, 1);
        gemm(midb, m2w2B, m2b2, a2f, a2f, a2b, DMODEL, 0);
    }

    // outer residual + MLP, then strided concat into d_out
    float* outp = (float*)d_out;
    add_kernel<<<EW_BLOCKS, 256, 0, stream>>>(x1, a1f, x1of);
    add_kernel<<<EW_BLOCKS, 256, 0, stream>>>(x2, a2f, x2of);

    ln(x1of, norm2_g, norm2_b, nullptr, ln1b);
    gemm(ln1b, f1B, f1b, nullptr, nullptr, midb, DMODEL, 2);
    gemm(midb, f2B, f2b, x1of, tmpf, nullptr, DMODEL, 0);
    writeout_kernel<<<EW_BLOCKS, 256, 0, stream>>>(tmpf, outp, 0);

    ln(x2of, norm2_g, norm2_b, nullptr, ln2b);
    gemm(ln2b, f1B, f1b, nullptr, nullptr, midb, DMODEL, 2);
    gemm(midb, f2B, f2b, x2of, tmpf, nullptr, DMODEL, 0);
    writeout_kernel<<<EW_BLOCKS, 256, 0, stream>>>(tmpf, outp, 1);
}
